// BiMambaBlock_6597069767166
// MI455X (gfx1250) — matrix operand order repaired and checked by reading, not compiled, not run on hardware
//
#include <hip/hip_runtime.h>

// ---------------------------------------------------------------------------
// BiMamba block for MI455X (gfx1250, wave32).
//  - Big GEMMs: 128x128 block tile, double-buffered LDS staging via
//    GLOBAL_LOAD_ASYNC_TO_LDS_B128 (ASYNCcnt) overlapped with
//    v_wmma_f32_16x16x32_bf16 compute; 8 waves * (2x4) register tiles.
//  - Small GEMM (N=80): direct-from-global register tiles.
//  - Conv / softplus / selective scan / silu in f32 VALU.
// ---------------------------------------------------------------------------

#define DMODEL 768
#define DINNER 1536
#define DSTATE 16
#define DCONV  4
#define DTRANK 48
#define DTPAD  64            // DTRANK padded to multiple of 32 for WMMA-K
#define BB     2
#define LL     1024
#define ROWS   (BB * LL)     // 2048
#define XPDIM  (DTRANK + 2 * DSTATE)  // 80

#define BM 128
#define BN 128
#define BK 32

typedef __attribute__((ext_vector_type(16))) __bf16          v16bf;
typedef __attribute__((ext_vector_type(8)))  float           v8f;
typedef __attribute__((ext_vector_type(8)))  unsigned short  ushort8;
typedef __attribute__((ext_vector_type(16))) unsigned short  ushort16;

__device__ __forceinline__ unsigned short f2bf(float f) {
  unsigned int u = __builtin_bit_cast(unsigned int, f);
  u += 0x7fffu + ((u >> 16) & 1u);       // round-to-nearest-even
  return (unsigned short)(u >> 16);
}

// Async copy of 16 bytes global -> LDS (per-lane), tracked by ASYNCcnt.
// ISA 10.2: within the LDS aperture, the low 32 bits of a
// generic pointer to __shared__ memory are the LDS byte offset.
__device__ __forceinline__ void async_copy_b128(void* lds_ptr, const void* gptr) {
  unsigned loff = (unsigned)(unsigned long long)lds_ptr;
  asm volatile("global_load_async_to_lds_b128 %0, %1, off"
               :: "v"(loff), "v"(gptr) : "memory");
}

__device__ __forceinline__ v16bf make_a_frag(const unsigned short* p_lo,
                                             const unsigned short* p_hi) {
  ushort8 alo = *(const ushort8*)p_lo;
  ushort8 ahi = *(const ushort8*)p_hi;
  ushort16 ua;
#pragma unroll
  for (int j = 0; j < 8; ++j) { ua[j] = alo[j]; ua[j + 8] = ahi[j]; }
  return __builtin_bit_cast(v16bf, ua);
}

// ---------------------------------------------------------------------------
// LDS-staged GEMM: C[M,N] = A[M,K] * W[N,K]^T  (bf16 in, f32 out/accum).
// Requires M%128==0, N%128==0, K%32==0. Block = 256 threads = 8 waves (4Mx2N),
// each wave owns a 32x64 register tile (2x4 wmma accumulators). Double-
// buffered panels: async copies for tile k+1 overlap wmma on tile k.
// ---------------------------------------------------------------------------
__global__ void __launch_bounds__(256) wmma_gemm_lds(
    const unsigned short* __restrict__ A,
    const unsigned short* __restrict__ W,
    float* __restrict__ C,
    int M, int N, int K, int accumulate)
{
  __shared__ __attribute__((aligned(32))) unsigned short sA[2][BM * BK];  // 16 KB
  __shared__ __attribute__((aligned(32))) unsigned short sB[2][BN * BK];  // 16 KB

  const int tid    = threadIdx.x;
  const int wid    = tid >> 5;         // 0..7
  const int lane   = tid & 31;
  const int wm     = wid >> 1;         // 0..3  : M sub-block (32 rows)
  const int wn     = wid & 1;          // 0..1  : N sub-block (64 cols)
  const int tilesN = N / BN;
  const int bm     = blockIdx.x / tilesN;
  const int bn     = blockIdx.x - bm * tilesN;
  const int r      = lane & 15;
  const int hi     = lane >> 4;

  // Per-thread copy coordinates: panels are 512 b128 chunks, 2 per thread.
  const int row0 = tid >> 2,          cc0 = tid & 3;
  const int row1 = (tid + 256) >> 2,  cc1 = (tid + 256) & 3;

  auto stage = [&](int buf, int k0) {
    async_copy_b128(&sA[buf][row0 * BK + cc0 * 8],
                    A + (size_t)(bm * BM + row0) * K + k0 + cc0 * 8);
    async_copy_b128(&sB[buf][row0 * BK + cc0 * 8],
                    W + (size_t)(bn * BN + row0) * K + k0 + cc0 * 8);
    async_copy_b128(&sA[buf][row1 * BK + cc1 * 8],
                    A + (size_t)(bm * BM + row1) * K + k0 + cc1 * 8);
    async_copy_b128(&sB[buf][row1 * BK + cc1 * 8],
                    W + (size_t)(bn * BN + row1) * K + k0 + cc1 * 8);
  };

  v8f acc[2][4] = {};
  const int nK = K / BK;

  stage(0, 0);
  for (int kt = 0; kt < nK; ++kt) {
    const int cur = kt & 1;
    __syncthreads();                 // readers of buf cur^1 (iter kt-1) done
    if (kt + 1 < nK) {
      stage(cur ^ 1, (kt + 1) * BK); // prefetch next tile into other buffer
      // async loads complete in order: <=4 outstanding => tile kt's are done
      asm volatile("s_wait_asynccnt 4" ::: "memory");
    } else {
      asm volatile("s_wait_asynccnt 0" ::: "memory");
    }
    __syncthreads();                 // all waves' tile-kt panels visible

    const unsigned short* pA = sA[cur];
    const unsigned short* pB = sB[cur];
#pragma unroll
    for (int i = 0; i < 2; ++i) {
      const int am = wm * 32 + i * 16 + r;     // local A row
      v16bf av = make_a_frag(&pA[am * BK + 8 * hi],
                             &pA[am * BK + 16 + 8 * hi]);
#pragma unroll
      for (int j = 0; j < 4; ++j) {
        const int bnr = wn * 64 + j * 16 + r;  // local B (=W) row
        v16bf bv = make_a_frag(&pB[bnr * BK + 8 * hi],
                               &pB[bnr * BK + 16 + 8 * hi]);
        acc[i][j] = __builtin_amdgcn_wmma_f32_16x16x32_bf16(
            false, av, false, bv, (short)0, acc[i][j], false, false);
      }
    }
  }

#pragma unroll
  for (int i = 0; i < 2; ++i) {
    const int mrow = bm * BM + wm * 32 + i * 16 + 8 * hi;
#pragma unroll
    for (int j = 0; j < 4; ++j) {
      const int ncol = bn * BN + wn * 64 + j * 16 + r;
      float* crow = C + (size_t)mrow * N + ncol;
      if (accumulate) {
        float old[8];
#pragma unroll
        for (int v = 0; v < 8; ++v) old[v] = crow[(size_t)v * N];
#pragma unroll
        for (int v = 0; v < 8; ++v) crow[(size_t)v * N] = old[v] + acc[i][j][v];
      } else {
#pragma unroll
        for (int v = 0; v < 8; ++v) crow[(size_t)v * N] = acc[i][j][v];
      }
    }
  }
}

// ---------------------------------------------------------------------------
// Simple register-tile GEMM (one wave = one 16x16 tile) for N not mult of 128.
// ---------------------------------------------------------------------------
__global__ void __launch_bounds__(256) wmma_gemm_bf16(
    const unsigned short* __restrict__ A,
    const unsigned short* __restrict__ W,
    float* __restrict__ C,
    int M, int N, int K, int accumulate)
{
  const int wave   = blockIdx.x * 8 + (threadIdx.x >> 5);
  const int lane   = threadIdx.x & 31;
  const int tilesN = N >> 4;
  const int tm     = wave / tilesN;
  const int tn     = wave - tm * tilesN;
  const int r      = lane & 15;
  const int hi     = lane >> 4;

  const unsigned short* arow = A + (size_t)(tm * 16 + r) * K;
  const unsigned short* brow = W + (size_t)(tn * 16 + r) * K;

  v8f acc = {};
  for (int k0 = 0; k0 < K; k0 += 32) {
    v16bf av = make_a_frag(arow + k0 + 8 * hi, arow + k0 + 16 + 8 * hi);
    v16bf bv = make_a_frag(brow + k0 + 8 * hi, brow + k0 + 16 + 8 * hi);
    acc = __builtin_amdgcn_wmma_f32_16x16x32_bf16(false, av, false, bv,
                                                  (short)0, acc, false, false);
  }
  float* crow = C + (size_t)(tm * 16 + 8 * hi) * N + tn * 16 + r;
  if (accumulate) {
    float old[8];
#pragma unroll
    for (int v = 0; v < 8; ++v) old[v] = crow[(size_t)v * N];
#pragma unroll
    for (int v = 0; v < 8; ++v) crow[(size_t)v * N] = old[v] + acc[v];
  } else {
#pragma unroll
    for (int v = 0; v < 8; ++v) crow[(size_t)v * N] = acc[v];
  }
}

// ---------------------------------------------------------------------------
// LayerNorm; also seeds out = x (residual base) and bf16 copy of xn.
// ---------------------------------------------------------------------------
__global__ void __launch_bounds__(256) ln_init_kernel(
    const float* __restrict__ x, const float* __restrict__ gamma,
    const float* __restrict__ beta, float* __restrict__ xn,
    unsigned short* __restrict__ xnb, float* __restrict__ out)
{
  const int row = blockIdx.x;
  const float* xr = x + (size_t)row * DMODEL;
  float s = 0.f, s2 = 0.f;
  for (int i = threadIdx.x; i < DMODEL; i += 256) {
    float v = xr[i]; s += v; s2 += v * v;
  }
  __shared__ float sh[2][8];
#pragma unroll
  for (int o = 16; o > 0; o >>= 1) {
    s += __shfl_xor(s, o, 32); s2 += __shfl_xor(s2, o, 32);
  }
  if ((threadIdx.x & 31) == 0) { sh[0][threadIdx.x >> 5] = s; sh[1][threadIdx.x >> 5] = s2; }
  __syncthreads();
  s = 0.f; s2 = 0.f;
#pragma unroll
  for (int w = 0; w < 8; ++w) { s += sh[0][w]; s2 += sh[1][w]; }
  const float mu  = s  * (1.f / DMODEL);
  const float var = s2 * (1.f / DMODEL) - mu * mu;
  const float rs  = rsqrtf(var + 1e-5f);
  for (int i = threadIdx.x; i < DMODEL; i += 256) {
    float v = xr[i];
    float y = (v - mu) * rs * gamma[i] + beta[i];
    xn [(size_t)row * DMODEL + i] = y;
    xnb[(size_t)row * DMODEL + i] = f2bf(y);
    out[(size_t)row * DMODEL + i] = v;
  }
}

__global__ void __launch_bounds__(256) f2bf_kernel(
    const float* __restrict__ s, unsigned short* __restrict__ d, int n)
{
  int i = blockIdx.x * 256 + threadIdx.x;
  if (i < n) d[i] = f2bf(s[i]);
}

// dt_w [DINNER,48] -> bf16 [DINNER,64] zero-padded
__global__ void __launch_bounds__(256) dtw_pad_kernel(
    const float* __restrict__ w, unsigned short* __restrict__ wb)
{
  int idx = blockIdx.x * 256 + threadIdx.x;     // DINNER*DTPAD
  int k = idx % DTPAD, d = idx / DTPAD;
  float v = (k < DTRANK) ? w[(size_t)d * DTRANK + k] : 0.f;
  wb[idx] = f2bf(v);
}

// x_dbl cols [0..47] -> bf16 [ROWS,64] zero-padded
__global__ void __launch_bounds__(256) dtpad_kernel(
    const float* __restrict__ xdbl, unsigned short* __restrict__ dtp)
{
  int idx = blockIdx.x * 256 + threadIdx.x;     // ROWS*DTPAD
  int k = idx % DTPAD, row = idx / DTPAD;
  float v = (k < DTRANK) ? xdbl[(size_t)row * XPDIM + k] : 0.f;
  dtp[idx] = f2bf(v);
}

// Depthwise conv (anti-causal taps when rev: flip∘conv∘flip) then SiLU.
__global__ void __launch_bounds__(256) conv_silu_kernel(
    const float* __restrict__ xz, const float* __restrict__ conv_w,
    const float* __restrict__ conv_b, float* __restrict__ xc,
    unsigned short* __restrict__ xcb, int rev)
{
  int idx = blockIdx.x * 256 + threadIdx.x;
  if (idx >= ROWS * DINNER) return;
  int d  = idx % DINNER;
  int rl = idx / DINNER;
  int l  = rl % LL;
  int b  = rl / LL;
  const float* w = conv_w + d * DCONV;
  float acc = conv_b[d];
  if (!rev) {
#pragma unroll
    for (int k = 0; k < DCONV; ++k) {
      int ls = l - (DCONV - 1) + k;
      if (ls >= 0) acc += w[k] * xz[((size_t)(b * LL + ls)) * (2 * DINNER) + d];
    }
  } else {
#pragma unroll
    for (int k = 0; k < DCONV; ++k) {
      int ls = l + (DCONV - 1) - k;
      if (ls < LL) acc += w[k] * xz[((size_t)(b * LL + ls)) * (2 * DINNER) + d];
    }
  }
  float v = acc / (1.f + __expf(-acc));    // silu
  xc [idx] = v;
  xcb[idx] = f2bf(v);
}

// delta = softplus(raw + dt_b[d])  (in place)
__global__ void __launch_bounds__(256) softplus_kernel(
    float* __restrict__ delta, const float* __restrict__ dt_b)
{
  int i = blockIdx.x * 256 + threadIdx.x;       // ROWS*DINNER
  int d = i % DINNER;
  float v = delta[i] + dt_b[d];
  delta[i] = (v > 0.f) ? (v + log1pf(__expf(-v))) : log1pf(__expf(v));
}

// Selective scan: one thread per (b, channel); 16 states in registers.
__global__ void __launch_bounds__(256) scan_kernel(
    const float* __restrict__ delta, const float* __restrict__ xc,
    const float* __restrict__ xdbl,  const float* __restrict__ xz,
    const float* __restrict__ A_log, const float* __restrict__ Dp,
    unsigned short* __restrict__ yb, int rev)
{
  int t = blockIdx.x * 256 + threadIdx.x;
  if (t >= BB * DINNER) return;
  int b = t / DINNER, d = t % DINNER;
  float A[DSTATE], h[DSTATE];
#pragma unroll
  for (int n = 0; n < DSTATE; ++n) {
    A[n] = -__expf(A_log[d * DSTATE + n]);
    h[n] = 0.f;
  }
  const float Dpd = Dp[d];
  for (int s = 0; s < LL; ++s) {
    int l = rev ? (LL - 1 - s) : s;
    size_t row = (size_t)b * LL + l;
    float dl = delta[row * DINNER + d];
    float xv = xc  [row * DINNER + d];
    const float* bc = xdbl + row * XPDIM + DTRANK;   // B_t [16]
    const float* cc = bc + DSTATE;                   // C_t [16]
    float dx = dl * xv;
    float y  = 0.f;
#pragma unroll
    for (int n = 0; n < DSTATE; ++n) {
      h[n] = __expf(dl * A[n]) * h[n] + dx * bc[n];
      y   += h[n] * cc[n];
    }
    y += xv * Dpd;
    float z = xz[row * (2 * DINNER) + DINNER + d];
    y *= z / (1.f + __expf(-z));                     // gate: silu(z)
    yb[row * DINNER + d] = f2bf(y);
  }
}

// ---------------------------------------------------------------------------
extern "C" void kernel_launch(void* const* d_in, const int* in_sizes, int n_in,
                              void* d_out, int out_size, void* d_ws, size_t ws_size,
                              hipStream_t stream) {
  (void)in_sizes; (void)n_in; (void)out_size; (void)ws_size;
  const float* x     = (const float*)d_in[0];
  const float* gamma = (const float*)d_in[1];
  const float* beta  = (const float*)d_in[2];
  float* out = (float*)d_out;

  char* ws = (char*)d_ws;
  size_t off = 0;
  auto take = [&](size_t bytes) -> char* {
    char* p = ws + off; off += (bytes + 255) & ~(size_t)255; return p;
  };
  float*          xn    = (float*)         take((size_t)ROWS * DMODEL * 4);
  unsigned short* xnb   = (unsigned short*)take((size_t)ROWS * DMODEL * 2);
  unsigned short* w_in  = (unsigned short*)take((size_t)2 * DINNER * DMODEL * 2);
  unsigned short* w_xp  = (unsigned short*)take((size_t)XPDIM * DINNER * 2);
  unsigned short* w_dt  = (unsigned short*)take((size_t)DINNER * DTPAD * 2);
  unsigned short* w_out = (unsigned short*)take((size_t)DMODEL * DINNER * 2);
  float*          xzb   = (float*)         take((size_t)ROWS * 2 * DINNER * 4);
  float*          xc    = (float*)         take((size_t)ROWS * DINNER * 4);
  unsigned short* xcb   = (unsigned short*)take((size_t)ROWS * DINNER * 2);
  float*          xdbl  = (float*)         take((size_t)ROWS * XPDIM * 4);
  unsigned short* dtp   = (unsigned short*)take((size_t)ROWS * DTPAD * 2);
  float*          dlt   = (float*)         take((size_t)ROWS * DINNER * 4);
  unsigned short* yb    = (unsigned short*)take((size_t)ROWS * DINNER * 2);
  (void)xn;

  ln_init_kernel<<<ROWS, 256, 0, stream>>>(x, gamma, beta, xn, xnb, out);

  for (int dir = 0; dir < 2; ++dir) {
    int bi = (dir == 0) ? 3 : 12;
    const float* in_w   = (const float*)d_in[bi + 0];
    const float* conv_w = (const float*)d_in[bi + 1];
    const float* conv_b = (const float*)d_in[bi + 2];
    const float* xp_w   = (const float*)d_in[bi + 3];
    const float* dt_w   = (const float*)d_in[bi + 4];
    const float* dt_b   = (const float*)d_in[bi + 5];
    const float* A_log  = (const float*)d_in[bi + 6];
    const float* Dp     = (const float*)d_in[bi + 7];
    const float* out_w  = (const float*)d_in[bi + 8];

    // weight staging -> bf16
    f2bf_kernel<<<(2 * DINNER * DMODEL) / 256, 256, 0, stream>>>(in_w, w_in, 2 * DINNER * DMODEL);
    f2bf_kernel<<<(XPDIM * DINNER) / 256, 256, 0, stream>>>(xp_w, w_xp, XPDIM * DINNER);
    dtw_pad_kernel<<<(DINNER * DTPAD) / 256, 256, 0, stream>>>(dt_w, w_dt);
    f2bf_kernel<<<(DMODEL * DINNER) / 256, 256, 0, stream>>>(out_w, w_out, DMODEL * DINNER);

    // xz = xn @ in_w^T        [2048 x 3072], K=768
    wmma_gemm_lds<<<(ROWS / BM) * (2 * DINNER / BN), 256, 0, stream>>>(
        xnb, w_in, xzb, ROWS, 2 * DINNER, DMODEL, 0);
    // depthwise conv + silu   -> xc
    conv_silu_kernel<<<(ROWS * DINNER) / 256, 256, 0, stream>>>(
        xzb, conv_w, conv_b, xc, xcb, dir);
    // x_dbl = xc @ xp_w^T     [2048 x 80], K=1536  (N not mult of 128)
    wmma_gemm_bf16<<<(ROWS / 16) * (XPDIM / 16) / 8, 256, 0, stream>>>(
        xcb, w_xp, xdbl, ROWS, XPDIM, DINNER, 0);
    // dt (padded to 64) -> bf16
    dtpad_kernel<<<(ROWS * DTPAD) / 256, 256, 0, stream>>>(xdbl, dtp);
    // delta_raw = dt @ dt_w^T [2048 x 1536], K=64
    wmma_gemm_lds<<<(ROWS / BM) * (DINNER / BN), 256, 0, stream>>>(
        dtp, w_dt, dlt, ROWS, DINNER, DTPAD, 0);
    // delta = softplus(raw + dt_b)
    softplus_kernel<<<(ROWS * DINNER) / 256, 256, 0, stream>>>(dlt, dt_b);
    // selective scan + D skip + silu(z) gate -> y (bf16)
    scan_kernel<<<(BB * DINNER + 255) / 256, 256, 0, stream>>>(
        dlt, xc, xdbl, xzb, A_log, Dp, yb, dir);
    // out += y @ out_w^T      [2048 x 768], K=1536
    wmma_gemm_lds<<<(ROWS / BM) * (DMODEL / BN), 256, 0, stream>>>(
        yb, w_out, out, ROWS, DMODEL, DINNER, 1);
  }
}
